// GKT_50276887167511
// MI455X (gfx1250) — matrix operand order repaired and checked by reading, not compiled, not run on hardware
//
#include <hip/hip_runtime.h>
#include <hip/hip_bf16.h>

typedef __attribute__((ext_vector_type(16))) _Float16 v16h;
typedef __attribute__((ext_vector_type(8)))  float    v8f;

#define CN 1024
#define HH 64
#define EE 64
#define DD 128
#define BB 256

// f16-element offsets of packed WMMA B-fragments inside d_ws
#define OFF_FN0W1 0        // K=256,N=64 -> 32 frags * 512
#define OFF_FN1W1 16384
#define OFF_FN0W2 32768    // K=64,N=64 -> 8 frags
#define OFF_FN1W2 36864
#define OFF_ERASE 40960
#define OFF_ADD   45056
#define OFF_WIH   49152    // K=64,N=192 -> 24 frags
#define OFF_WHH   61440
#define WS_HALF_TOTAL 73728

// ---------------------------------------------------------------------------
// Pack a row-major (K,N) fp32 weight into f16 WMMA B-fragments:
// frag (kt,nt): lane l covers n = nt*16 + (l&15), k = kt*32 + (l>>4)*8 + (j&7) + (j>>3)*16,
// stored as 512 contiguous halves (lane*16 + j) so the GEMM kernel loads each
// fragment with two b128 loads.
// ---------------------------------------------------------------------------
__global__ void pack_wfrag(const float* __restrict__ src, _Float16* __restrict__ dst,
                           int K, int N) {
  int idx = blockIdx.x * blockDim.x + threadIdx.x;
  int NT = N >> 4;
  int total = (K >> 5) * NT * 512;
  if (idx >= total) return;
  int frag = idx >> 9;
  int lane = (idx >> 4) & 31;
  int j    = idx & 15;
  int kt = frag / NT, nt = frag - kt * NT;
  int k = kt * 32 + ((lane >> 4) & 1) * 8 + (j & 7) + (j >> 3) * 16;
  int n = nt * 16 + (lane & 15);
  dst[idx] = (_Float16)src[k * N + n];
}

__device__ __forceinline__ v8f wmma16(v16h a, v16h b, v8f c) {
  return __builtin_amdgcn_wmma_f32_16x16x32_f16(false, a, false, b, (short)0, c,
                                                false, false);
}

// Build a 16x32 f16 A-fragment from an LDS row-major tile.
// A layout (ISA 7.12.2): lanes0-15 m=0..15, VGPRj holds K pairs
// {0,1},{2,3},{4,5},{6,7},{16,17},{18,19},{20,21},{22,23}; lanes16-31 +8.
__device__ __forceinline__ v16h load_afrag(const _Float16* base, int rowStride,
                                           int kbase, int lane) {
  int m = lane & 15;
  int koff = ((lane >> 4) & 1) * 8;
  union { uint32_t u[8]; v16h v; } a;
  const _Float16* rp = base + m * rowStride + kbase + koff;
#pragma unroll
  for (int j = 0; j < 8; ++j) {
    int kk = ((j & 3) * 2) + ((j >> 2) * 16);
    a.u[j] = *reinterpret_cast<const uint32_t*>(rp + kk);
  }
  return a.v;
}

__device__ __forceinline__ v16h load_bfrag(const _Float16* base, int frag, int lane) {
  return *reinterpret_cast<const v16h*>(base + frag * 512 + lane * 16);
}

__device__ __forceinline__ float sigmoidf_(float x) { return 1.0f / (1.0f + __expf(-x)); }

// ---------------------------------------------------------------------------
// Fused forward. Block = 128 threads (4 wave32). blockIdx.y = batch b,
// blockIdx.x = 64-row chunk of CN. Each wave owns a 16-row tile.
// ---------------------------------------------------------------------------
__global__ __launch_bounds__(128) void gkt_forward(
    const int* __restrict__ xt, const int* __restrict__ qt,
    const float* __restrict__ ht, const float* __restrict__ graph,
    const float* __restrict__ emb_x, const float* __restrict__ emb_c,
    const float* __restrict__ fs_W1, const float* __restrict__ fs_b1,
    const float* __restrict__ fs_W2, const float* __restrict__ fs_b2,
    const float* __restrict__ fn0_b1, const float* __restrict__ fn0_b2,
    const float* __restrict__ fn1_b1, const float* __restrict__ fn1_b2,
    const float* __restrict__ ea_weight,
    const float* __restrict__ erase_b, const float* __restrict__ add_b,
    const float* __restrict__ gru_bih, const float* __restrict__ gru_bhh,
    const float* __restrict__ pred_W, const float* __restrict__ pred_b,
    const _Float16* __restrict__ wf, float* __restrict__ out) {
  // LDS carve (59392 B):
  //  [0,32768)      sNeigh  : 4 waves * 16 rows * 256 f16  (self_ht|ht|emb)
  //  [32768,49152)  sH1a/sH1b (f16) -- aliased later by sF32 (f32 h_next)
  //  [49152,57344)  sMnext  : 4 * 16 * 64 f16
  //  [57344,59392)  misc floats
  __shared__ __align__(16) char smem[59392];
  _Float16* sNeigh = (_Float16*)(smem);
  _Float16* sH1a   = (_Float16*)(smem + 32768);
  _Float16* sH1b   = (_Float16*)(smem + 40960);
  float*    sF32   = (float*)(smem + 32768);
  _Float16* sMnext = (_Float16*)(smem + 49152);
  float* sMisc     = (float*)(smem + 57344);
  float* sSelfHt   = sMisc;         // 128
  float* sTmp      = sMisc + 128;   // 64
  float* sSelfFeat = sMisc + 192;   // 64
  float* sAdj      = sMisc + 256;   // 64
  float* sRadj     = sMisc + 320;   // 64
  float* sEa       = sMisc + 384;   // 64
  float* sPredW    = sMisc + 448;   // 64

  const int tid  = threadIdx.x;
  const int wave = tid >> 5;
  const int lane = tid & 31;
  const int b    = blockIdx.y;
  const int c0   = blockIdx.x * 64;

  const int  qv = qt[b];
  const bool qm = (qv != -1);
  const int  qs = qv < 0 ? 0 : (qv > CN - 1 ? CN - 1 : qv);
  const int  xv = xt[b];

  // ---- stage self_ht (D=128), predW ----
  if (tid < 64) {
    sSelfHt[tid] = ht[((size_t)b * CN + qs) * HH + tid];
    sPredW[tid]  = pred_W[tid];
  } else if (tid < 128) {
    int e = tid - 64;
    sSelfHt[64 + e] = qm ? emb_x[(size_t)xv * EE + e] : emb_c[(size_t)CN * EE + e];
  }
  __syncthreads();

  // ---- fs MLP (1x128 @ 128x64 -> relu -> 1x64 @ 64x64 -> relu), scalar ----
  if (tid < 64) {
    float acc = fs_b1[tid];
    for (int k = 0; k < DD; ++k) acc += sSelfHt[k] * fs_W1[k * HH + tid];
    sTmp[tid] = fmaxf(acc, 0.f);
  }
  __syncthreads();
  if (tid < 64) {
    float acc = fs_b2[tid];
    for (int k = 0; k < HH; ++k) acc += sTmp[k] * fs_W2[k * HH + tid];
    sSelfFeat[tid] = fmaxf(acc, 0.f);
    // adj / radj / ea for the 64 rows of this block
    int c = c0 + tid;
    sAdj[tid]  = graph[(size_t)qs * CN + c];
    sRadj[tid] = graph[(size_t)c * CN + qs];
    sEa[tid]   = ea_weight[c];
  }
  __syncthreads();

  // ---- per-wave: stage 16x256 neigh tile as f16 ----
  _Float16* sNeighW = sNeigh + wave * 16 * 256;
  {
    int r = lane >> 1, hhf = lane & 1;
    int c = c0 + wave * 16 + r;
    _Float16* drow = sNeighW + r * 256;
    union { _Float16 h[2]; uint32_t u; } p;
    if (hhf == 0) {  // cols 0..127 : self_ht (broadcast)
#pragma unroll 4
      for (int i = 0; i < 128; i += 2) {
        p.h[0] = (_Float16)sSelfHt[i]; p.h[1] = (_Float16)sSelfHt[i + 1];
        *reinterpret_cast<uint32_t*>(drow + i) = p.u;
      }
    } else {         // cols 128..191 : ht row ; 192..255 : concept emb
      const float* hrow = ht + ((size_t)b * CN + c) * HH;
      const float* ep = qm ? ((c == qs) ? (emb_x + (size_t)xv * EE)
                                        : (emb_c + (size_t)c * EE))
                           : (emb_c + (size_t)CN * EE);
#pragma unroll 4
      for (int i = 0; i < 64; i += 2) {
        p.h[0] = (_Float16)hrow[i]; p.h[1] = (_Float16)hrow[i + 1];
        *reinterpret_cast<uint32_t*>(drow + 128 + i) = p.u;
        p.h[0] = (_Float16)ep[i]; p.h[1] = (_Float16)ep[i + 1];
        *reinterpret_cast<uint32_t*>(drow + 192 + i) = p.u;
      }
    }
  }
  // per-wave LDS region, same-wave in-order LDS => no barrier needed here

  const int hi = lane >> 4;
  const int n_ = lane & 15;
  const v8f zero8 = {};
  float mn[4][8];  // m_next in C-layout: row m = v + hi*8, col = nt*16 + n_

  if (qm) {
    // ===== GEMM1: neigh(16x256) @ W1(256x64), branches fn0 & fn1 =====
    v8f acc0[4], acc1[4];
#pragma unroll
    for (int nt = 0; nt < 4; ++nt) { acc0[nt] = zero8; acc1[nt] = zero8; }
#pragma unroll
    for (int kt = 0; kt < 8; ++kt) {
      v16h A = load_afrag(sNeighW, 256, kt * 32, lane);
#pragma unroll
      for (int nt = 0; nt < 4; ++nt) {
        acc0[nt] = wmma16(A, load_bfrag(wf + OFF_FN0W1, kt * 4 + nt, lane), acc0[nt]);
        acc1[nt] = wmma16(A, load_bfrag(wf + OFF_FN1W1, kt * 4 + nt, lane), acc1[nt]);
      }
    }
    // bias + relu -> f16 hidden tiles
    _Float16* h1a = sH1a + wave * 1024;
    _Float16* h1b = sH1b + wave * 1024;
#pragma unroll
    for (int nt = 0; nt < 4; ++nt) {
      float ba = fn0_b1[nt * 16 + n_], bb = fn1_b1[nt * 16 + n_];
#pragma unroll
      for (int v = 0; v < 8; ++v) {
        int m = v + hi * 8;
        h1a[m * 64 + nt * 16 + n_] = (_Float16)fmaxf(acc0[nt][v] + ba, 0.f);
        h1b[m * 64 + nt * 16 + n_] = (_Float16)fmaxf(acc1[nt][v] + bb, 0.f);
      }
    }
    // ===== GEMM2: h1(16x64) @ W2(64x64) =====
#pragma unroll
    for (int nt = 0; nt < 4; ++nt) { acc0[nt] = zero8; acc1[nt] = zero8; }
#pragma unroll
    for (int kt = 0; kt < 2; ++kt) {
      v16h A0 = load_afrag(h1a, 64, kt * 32, lane);
      v16h A1 = load_afrag(h1b, 64, kt * 32, lane);
#pragma unroll
      for (int nt = 0; nt < 4; ++nt) {
        acc0[nt] = wmma16(A0, load_bfrag(wf + OFF_FN0W2, kt * 4 + nt, lane), acc0[nt]);
        acc1[nt] = wmma16(A1, load_bfrag(wf + OFF_FN1W2, kt * 4 + nt, lane), acc1[nt]);
      }
    }
    // combine: neigh_feat = adj*relu(f0)+radj*relu(f1); at_q -> self_feat
#pragma unroll
    for (int nt = 0; nt < 4; ++nt) {
      float b20 = fn0_b2[nt * 16 + n_], b21 = fn1_b2[nt * 16 + n_];
#pragma unroll
      for (int v = 0; v < 8; ++v) {
        int m = v + hi * 8;
        float f0 = fmaxf(acc0[nt][v] + b20, 0.f);
        float f1 = fmaxf(acc1[nt][v] + b21, 0.f);
        float val = sAdj[wave * 16 + m] * f0 + sRadj[wave * 16 + m] * f1;
        if (c0 + wave * 16 + m == qs) val = sSelfFeat[nt * 16 + n_];
        mn[nt][v] = val;
      }
    }
    // stage m_next for erase/add GEMMs
    _Float16* mW = sMnext + wave * 1024;
#pragma unroll
    for (int nt = 0; nt < 4; ++nt)
#pragma unroll
      for (int v = 0; v < 8; ++v) {
        int m = v + hi * 8;
        mW[m * 64 + nt * 16 + n_] = (_Float16)mn[nt][v];
      }
    // ===== erase/add gates =====
    v8f eacc[4], aacc[4];
#pragma unroll
    for (int nt = 0; nt < 4; ++nt) { eacc[nt] = zero8; aacc[nt] = zero8; }
#pragma unroll
    for (int kt = 0; kt < 2; ++kt) {
      v16h A = load_afrag(mW, 64, kt * 32, lane);
#pragma unroll
      for (int nt = 0; nt < 4; ++nt) {
        eacc[nt] = wmma16(A, load_bfrag(wf + OFF_ERASE, kt * 4 + nt, lane), eacc[nt]);
        aacc[nt] = wmma16(A, load_bfrag(wf + OFF_ADD,   kt * 4 + nt, lane), aacc[nt]);
      }
    }
#pragma unroll
    for (int nt = 0; nt < 4; ++nt) {
      float be = erase_b[nt * 16 + n_], bd = add_b[nt * 16 + n_];
#pragma unroll
      for (int v = 0; v < 8; ++v) {
        int m = v + hi * 8;
        float eg = sigmoidf_(eacc[nt][v] + be);
        float af = tanhf(aacc[nt][v] + bd);
        float wv = sEa[wave * 16 + m];
        float x = mn[nt][v];
        mn[nt][v] = x - wv * eg * x + wv * af;
      }
    }
    // restage gated m_next for GRU input GEMM
#pragma unroll
    for (int nt = 0; nt < 4; ++nt)
#pragma unroll
      for (int v = 0; v < 8; ++v) {
        int m = v + hi * 8;
        mW[m * 64 + nt * 16 + n_] = (_Float16)mn[nt][v];
      }
    // ===== GRU: gi = m_next @ Wih(64x192), gh = ht @ Whh(64x192) =====
    float rr[4][8], zz[4][8];
#pragma unroll
    for (int g = 0; g < 3; ++g) {  // gate groups: r, z, n
      v8f gi[4], gh[4];
#pragma unroll
      for (int j = 0; j < 4; ++j) { gi[j] = zero8; gh[j] = zero8; }
#pragma unroll
      for (int kt = 0; kt < 2; ++kt) {
        v16h Am = load_afrag(mW, 64, kt * 32, lane);
        v16h Ah = load_afrag(sNeighW + 128, 256, kt * 32, lane);  // ht cols
#pragma unroll
        for (int j = 0; j < 4; ++j) {
          int nt = g * 4 + j;
          gi[j] = wmma16(Am, load_bfrag(wf + OFF_WIH, kt * 12 + nt, lane), gi[j]);
          gh[j] = wmma16(Ah, load_bfrag(wf + OFF_WHH, kt * 12 + nt, lane), gh[j]);
        }
      }
#pragma unroll
      for (int j = 0; j < 4; ++j) {
        int col = (g * 4 + j) * 16 + n_;
        float bi = gru_bih[col], bh = gru_bhh[col];
#pragma unroll
        for (int v = 0; v < 8; ++v) {
          float x = gi[j][v] + bi;
          float y = gh[j][v] + bh;
          if (g == 0) rr[j][v] = sigmoidf_(x + y);
          else if (g == 1) zz[j][v] = sigmoidf_(x + y);
          else {
            int m = v + hi * 8;
            float nn  = tanhf(x + rr[j][v] * y);
            float htv = (float)sNeighW[m * 256 + 128 + j * 16 + n_];
            mn[j][v] = (1.f - zz[j][v]) * nn + zz[j][v] * htv;  // h_next
          }
        }
      }
    }
  } else {
    // qt masked: h_next = ht
#pragma unroll
    for (int nt = 0; nt < 4; ++nt)
#pragma unroll
      for (int v = 0; v < 8; ++v) {
        int m = v + hi * 8;
        mn[nt][v] = (float)sNeighW[m * 256 + 128 + nt * 16 + n_];
      }
  }

  __syncthreads();  // all waves done reading sH1 region before aliasing as sF32

  float* fW = sF32 + wave * 1024;
#pragma unroll
  for (int nt = 0; nt < 4; ++nt)
#pragma unroll
    for (int v = 0; v < 8; ++v) {
      int m = v + hi * 8;
      fW[m * 64 + nt * 16 + n_] = mn[nt][v];
    }

  // ---- pred head: y = h_next @ pred_W + pred_b ----
  if (lane < 16) {
    float acc = pred_b[0];
#pragma unroll 8
    for (int n = 0; n < 64; ++n) acc += fW[lane * 64 + n] * sPredW[n];
    float yv = qm ? sigmoidf_(acc) : acc;
    out[(size_t)b * CN + c0 + wave * 16 + lane] = yv;
  }
}

extern "C" void kernel_launch(void* const* d_in, const int* in_sizes, int n_in,
                              void* d_out, int out_size, void* d_ws, size_t ws_size,
                              hipStream_t stream) {
  const int*   xt      = (const int*)d_in[0];
  const int*   qt      = (const int*)d_in[1];
  const float* ht      = (const float*)d_in[2];
  const float* graph   = (const float*)d_in[3];
  const float* emb_x   = (const float*)d_in[4];
  const float* emb_c   = (const float*)d_in[5];
  const float* fs_W1   = (const float*)d_in[6];
  const float* fs_b1   = (const float*)d_in[7];
  const float* fs_W2   = (const float*)d_in[8];
  const float* fs_b2   = (const float*)d_in[9];
  const float* fn0_W1  = (const float*)d_in[10];
  const float* fn0_b1  = (const float*)d_in[11];
  const float* fn0_W2  = (const float*)d_in[12];
  const float* fn0_b2  = (const float*)d_in[13];
  const float* fn1_W1  = (const float*)d_in[14];
  const float* fn1_b1  = (const float*)d_in[15];
  const float* fn1_W2  = (const float*)d_in[16];
  const float* fn1_b2  = (const float*)d_in[17];
  const float* ea_w    = (const float*)d_in[18];
  const float* erase_W = (const float*)d_in[19];
  const float* erase_b = (const float*)d_in[20];
  const float* add_W   = (const float*)d_in[21];
  const float* add_b   = (const float*)d_in[22];
  const float* gru_Wih = (const float*)d_in[23];
  const float* gru_bih = (const float*)d_in[24];
  const float* gru_Whh = (const float*)d_in[25];
  const float* gru_bhh = (const float*)d_in[26];
  const float* pred_W  = (const float*)d_in[27];
  const float* pred_b  = (const float*)d_in[28];

  _Float16* wf = (_Float16*)d_ws;  // needs 147456 B of workspace
  auto pack = [&](const float* src, int off, int K, int N) {
    int total = (K / 32) * (N / 16) * 512;
    pack_wfrag<<<(total + 255) / 256, 256, 0, stream>>>(src, wf + off, K, N);
  };
  pack(fn0_W1, OFF_FN0W1, 256, 64);
  pack(fn1_W1, OFF_FN1W1, 256, 64);
  pack(fn0_W2, OFF_FN0W2, 64, 64);
  pack(fn1_W2, OFF_FN1W2, 64, 64);
  pack(erase_W, OFF_ERASE, 64, 64);
  pack(add_W,  OFF_ADD,   64, 64);
  pack(gru_Wih, OFF_WIH, 64, 192);
  pack(gru_Whh, OFF_WHH, 64, 192);

  dim3 grid(CN / 64, BB);
  gkt_forward<<<grid, 128, 0, stream>>>(
      xt, qt, ht, graph, emb_x, emb_c,
      fs_W1, fs_b1, fs_W2, fs_b2,
      fn0_b1, fn0_b2, fn1_b1, fn1_b2,
      ea_w, erase_b, add_b, gru_bih, gru_bhh, pred_W, pred_b,
      (const _Float16*)d_ws, (float*)d_out);
}
